// AutoInt_62938450755702
// MI455X (gfx1250) — matrix operand order repaired and checked by reading, not compiled, not run on hardware
//
#include <hip/hip_runtime.h>
#include <hip/hip_bf16.h>

typedef __attribute__((ext_vector_type(16))) __bf16 v16bf;
typedef __attribute__((ext_vector_type(8)))  float  v8f;

#define B_F   39      // features per batch row
#define E_DIM 64
#define U_DIM 256
#define H_NUM 8
#define DH    32
#define FP    48      // F padded to 3 M-tiles
#define KP    64      // K padded for attn P@V stage

// ---- dynamic LDS layout (bytes, all 16B aligned) ----
#define OFF_EMB   0         // 48*64  bf16 = 6144
#define OFF_Q     6144      // 48*256 bf16 = 24576
#define OFF_K     30720     // 48*256 bf16 = 24576
#define OFF_V     55296     // 64*256 bf16 = 32768 (rows 48..63 zero pad)
#define OFF_P     88064     // 8 heads * 48*64 bf16 = 49152
#define OFF_LOGIT 137216    // float
#define SMEM_BYTES 137232

__device__ __forceinline__ v8f wmma_bf16(v16bf a, v16bf b, v8f c) {
  // D = A(16x32) * B(32x16) + C, f32 accumulate
  return __builtin_amdgcn_wmma_f32_16x16x32_bf16(
      /*neg_a=*/false, a, /*neg_b=*/false, b,
      /*c_mod=*/(short)0, c, /*reuse_a=*/false, /*reuse_b=*/false);
}

// A-fragment (16x32 bf16, M x K) from LDS, ISA 7.12.2 layout:
// lane&15 = M; lanes 0-15 hold K {0..7,16..23}, lanes 16-31 hold K {8..15,24..31}
__device__ __forceinline__ v16bf load_a_frag(const __bf16* S, int ld, int mbase,
                                             int kbase, int lane) {
  v16bf a;
  const __bf16* row = S + (mbase + (lane & 15)) * ld + kbase;
  const int koff = (lane >> 4) << 3;
#pragma unroll
  for (int t = 0; t < 16; ++t) {
    int k = ((t < 8) ? t : t + 8) + koff;
    a[t] = row[k];
  }
  return a;
}

// B-fragment (32x16 bf16, K x N) from LDS: lane&15 = N; lanes 0-15 hold
// K {0..7,16..23}, lanes 16-31 hold K {8..15,24..31}
__device__ __forceinline__ v16bf load_b_frag_lds(const __bf16* T, int ld,
                                                 int kbase, int nbase, int lane) {
  v16bf b;
  const __bf16* col = T + kbase * ld + nbase + (lane & 15);
  const int koff = (lane >> 4) << 3;
#pragma unroll
  for (int t = 0; t < 16; ++t) b[t] = col[(((t < 8) ? t : t + 8) + koff) * ld];
  return b;
}

// B-fragment from global f32 weight matrix (convert to bf16 on the fly)
__device__ __forceinline__ v16bf load_b_frag_w(const float* T, int ld,
                                               int kbase, int nbase, int lane) {
  v16bf b;
  const float* col = T + kbase * ld + nbase + (lane & 15);
  const int koff = (lane >> 4) << 3;
#pragma unroll
  for (int t = 0; t < 16; ++t) b[t] = (__bf16)col[(((t < 8) ? t : t + 8) + koff) * ld];
  return b;
}

__global__ __launch_bounds__(256)
void autoint_fwd_kernel(const int* __restrict__ fid,
                        const float* __restrict__ emb_table,
                        const float* __restrict__ Wq, const float* __restrict__ bq,
                        const float* __restrict__ Wk, const float* __restrict__ bk,
                        const float* __restrict__ Wv, const float* __restrict__ bv,
                        const float* __restrict__ Wp, const float* __restrict__ bp,
                        float* __restrict__ out) {
  extern __shared__ char smem[];
  __bf16* s_emb = (__bf16*)(smem + OFF_EMB);            // [48][64]
  __bf16* s_q   = (__bf16*)(smem + OFF_Q);              // [48][256]
  __bf16* s_k   = (__bf16*)(smem + OFF_K);              // [48][256]
  __bf16* s_v   = (__bf16*)(smem + OFF_V);              // [64][256]
  __bf16* s_p   = (__bf16*)(smem + OFF_P);              // [8][48][64]
  float*  s_logit = (float*)(smem + OFF_LOGIT);

  const int b    = blockIdx.x;
  const int tid  = threadIdx.x;
  const int lane = tid & 31;
  const int h    = tid >> 5;        // wave == head (8 waves, 8 heads)
  const int cb0  = h * DH;          // this head's column base in U

  if (tid == 0) *s_logit = 0.0f;

  // warm L2 path for weight rows (emits global_prefetch_b8)
  if (tid < E_DIM) {
    __builtin_prefetch(Wq + tid * U_DIM, 0, 1);
    __builtin_prefetch(Wk + tid * U_DIM, 0, 1);
    __builtin_prefetch(Wv + tid * U_DIM, 0, 1);
  }

  // ---- Phase 0: gather embeddings -> LDS (bf16), rows 39..47 zero ----
  for (int i = tid; i < FP * E_DIM; i += 256) {
    int r = i >> 6, c = i & 63;
    float v = 0.0f;
    if (r < B_F) {
      int f = fid[b * B_F + r];
      v = emb_table[(long long)f * E_DIM + c];
    }
    s_emb[i] = (__bf16)v;
  }
  __syncthreads();

  // zero pad: V rows 48..63 (this head's cols) and P cols 48..63 (this head)
  __bf16* s_ph = s_p + h * (FP * KP);
  for (int i = lane; i < 16 * DH; i += 32) {
    s_v[(FP + (i >> 5)) * U_DIM + cb0 + (i & 31)] = (__bf16)0.0f;
  }
  for (int i = lane; i < FP * 16; i += 32) {
    s_ph[(i >> 4) * KP + FP + (i & 15)] = (__bf16)0.0f;
  }

  // ---- Phase 1: Q/K/V projections via WMMA (per-wave: own head's 32 cols) --
  v16bf ea[3][2];
#pragma unroll
  for (int mt = 0; mt < 3; ++mt)
#pragma unroll
    for (int kt = 0; kt < 2; ++kt)
      ea[mt][kt] = load_a_frag(s_emb, E_DIM, mt * 16, kt * 32, lane);

  const float*  Ws[3] = {Wq, Wk, Wv};
  const float*  bs[3] = {bq, bk, bv};
  __bf16*       Ds[3] = {s_q, s_k, s_v};
#pragma unroll
  for (int p = 0; p < 3; ++p) {
#pragma unroll
    for (int nt = 0; nt < 2; ++nt) {
      const int cb = cb0 + nt * 16;
      v16bf wb0 = load_b_frag_w(Ws[p], U_DIM, 0,  cb, lane);
      v16bf wb1 = load_b_frag_w(Ws[p], U_DIM, 32, cb, lane);
      const float bias = bs[p][cb + (lane & 15)];
#pragma unroll
      for (int mt = 0; mt < 3; ++mt) {
        v8f acc = {};
        acc = wmma_bf16(ea[mt][0], wb0, acc);
        acc = wmma_bf16(ea[mt][1], wb1, acc);
        // C layout: VGPR r -> row r (+8 for lanes 16..31), col = lane&15
#pragma unroll
        for (int r = 0; r < 8; ++r) {
          float x = fmaxf(acc[r] + bias, 0.0f);   // bias + ReLU
          int row = mt * 16 + r + ((lane >> 4) << 3);
          Ds[p][row * U_DIM + cb + (lane & 15)] = (__bf16)x;
        }
      }
    }
  }
  // per-head data is produced and consumed by the same wave: no block sync

  // ---- Phase 2: scores = Q Kt / sqrt(dh), softmax (per head) ----
  v16bf qa[3];
#pragma unroll
  for (int mt = 0; mt < 3; ++mt)
    qa[mt] = load_a_frag(s_q, U_DIM, mt * 16, cb0, lane);
  v16bf kb[3];
#pragma unroll
  for (int nt = 0; nt < 3; ++nt) {
    // B[k][n] = K[nt*16+n][cb0+k] -> lane&15 = n (feature), t -> k (dh, table order)
    v16bf bf;
    const __bf16* base = s_k + (nt * 16 + (lane & 15)) * U_DIM + cb0 + ((lane >> 4) << 3);
#pragma unroll
    for (int t = 0; t < 16; ++t) bf[t] = base[(t < 8) ? t : t + 8];
    kb[nt] = bf;
  }
  v8f sc[3][3];
#pragma unroll
  for (int mt = 0; mt < 3; ++mt)
#pragma unroll
    for (int nt = 0; nt < 3; ++nt) {
      v8f z = {};
      sc[mt][nt] = wmma_bf16(qa[mt], kb[nt], z);
    }

  const float scale = 0.17677669529663689f;  // 1/sqrt(32)
  const int c = lane & 15;
#pragma unroll
  for (int mt = 0; mt < 3; ++mt) {
#pragma unroll
    for (int r = 0; r < 8; ++r) {
      float v0 = sc[mt][0][r] * scale;
      float v1 = sc[mt][1][r] * scale;
      float v2 = (32 + c < B_F) ? sc[mt][2][r] * scale : -1e30f;  // col mask
      float m = fmaxf(v0, fmaxf(v1, v2));
#pragma unroll
      for (int off = 8; off >= 1; off >>= 1)
        m = fmaxf(m, __shfl_xor(m, off, 32));   // row lives in a 16-lane half
      float e0 = __expf(v0 - m), e1 = __expf(v1 - m), e2 = __expf(v2 - m);
      float s3 = e0 + e1 + e2;
#pragma unroll
      for (int off = 8; off >= 1; off >>= 1)
        s3 += __shfl_xor(s3, off, 32);
      float inv = 1.0f / s3;
      int row = mt * 16 + r + ((lane >> 4) << 3);
      __bf16* prow = s_ph + row * KP;
      prow[c]      = (__bf16)(e0 * inv);
      prow[16 + c] = (__bf16)(e1 * inv);
      prow[32 + c] = (__bf16)(e2 * inv);
    }
  }

  // ---- Phase 3: out = P @ V (K padded to 64 with zeros) ----
  v16bf pa[3][2];
#pragma unroll
  for (int mt = 0; mt < 3; ++mt)
#pragma unroll
    for (int kt = 0; kt < 2; ++kt)
      pa[mt][kt] = load_a_frag(s_ph, KP, mt * 16, kt * 32, lane);
  v16bf vb[2][2];
#pragma unroll
  for (int kt = 0; kt < 2; ++kt)
#pragma unroll
    for (int nt = 0; nt < 2; ++nt)
      vb[kt][nt] = load_b_frag_lds(s_v, U_DIM, kt * 32, cb0 + nt * 16, lane);

  // ---- Phase 4: ReLU + dot with Wp, block reduction ----
  float partial = 0.0f;
#pragma unroll
  for (int mt = 0; mt < 3; ++mt)
#pragma unroll
    for (int nt = 0; nt < 2; ++nt) {
      v8f o = {};
      o = wmma_bf16(pa[mt][0], vb[0][nt], o);
      o = wmma_bf16(pa[mt][1], vb[1][nt], o);
#pragma unroll
      for (int r = 0; r < 8; ++r) {
        int row = mt * 16 + r + ((lane >> 4) << 3);
        if (row < B_F) {
          int col = cb0 + nt * 16 + (lane & 15);
          float x = fmaxf(o[r], 0.0f);
          partial += x * Wp[row * U_DIM + col];
        }
      }
    }
#pragma unroll
  for (int off = 16; off >= 1; off >>= 1)
    partial += __shfl_xor(partial, off, 32);
  if (lane == 0) atomicAdd(s_logit, partial);   // ds_add_f32

  __syncthreads();
  if (tid == 0) out[b] = *s_logit + bp[0];
}

extern "C" void kernel_launch(void* const* d_in, const int* in_sizes, int n_in,
                              void* d_out, int out_size, void* d_ws, size_t ws_size,
                              hipStream_t stream) {
  (void)n_in; (void)d_ws; (void)ws_size; (void)out_size;
  const int*   fid = (const int*)d_in[0];
  const float* emb = (const float*)d_in[1];
  const float* Wq  = (const float*)d_in[2];
  const float* bq  = (const float*)d_in[3];
  const float* Wk  = (const float*)d_in[4];
  const float* bk  = (const float*)d_in[5];
  const float* Wv  = (const float*)d_in[6];
  const float* bv  = (const float*)d_in[7];
  const float* Wp  = (const float*)d_in[8];
  const float* bp  = (const float*)d_in[9];
  float* out = (float*)d_out;

  const int B = in_sizes[0] / B_F;  // 4096

  // >64KB dynamic LDS opt-in (idempotent; no alloc/sync — graph-capture safe)
  hipFuncSetAttribute((const void*)autoint_fwd_kernel,
                      hipFuncAttributeMaxDynamicSharedMemorySize, SMEM_BYTES);

  autoint_fwd_kernel<<<B, 256, SMEM_BYTES, stream>>>(
      fid, emb, Wq, bq, Wk, bk, Wv, bv, Wp, bp, out);
}
